// SimpleGNN_13142599925774
// MI455X (gfx1250) — compile-verified, matrix operand order checked by reading, not run on hardware
//
#include <hip/hip_runtime.h>
#include <hip/hip_bf16.h>

#define N_NODES 100000   // multiple of 16 -> GEMM row tiles never partial
#define N_EDGES 1600000
#define HIDDEN  128
#define N_SUBG  2048     // multiple of 16
#define MAX_SUBG 64
#define OUT_DIM 16

typedef __attribute__((ext_vector_type(16))) __bf16 v16bf;
typedef __attribute__((ext_vector_type(8)))  float v8f;

union Frag { v16bf v; uint4 q[2]; };
union Pack8 { __bf16 b[8]; uint4 q; };

// ---------------------------------------------------------------------------
// WMMA GEMM: C[M x NCOLS] = A[M x 128] @ W[128 x NCOLS] (+bias, optional ReLU)
// A is bf16 row-major [M][128]; Wt is bf16 pre-transposed [NCOLS][128].
// Wt is staged in LDS; fragments are pure 128-bit loads (no per-element VALU).
// One wave computes a 16-row strip across all NCOLS columns; 8 waves/block.
// REQUIRES: M % 16 == 0 (holds for N_NODES and N_SUBG) -> unguarded stores.
// ---------------------------------------------------------------------------
template<int NCOLS, bool RELU, bool HASBIAS, bool BF16OUT>
__global__ __launch_bounds__(256)
void gemm_wmma_bf16(const __bf16* __restrict__ A, const __bf16* __restrict__ Wt,
                    const float* __restrict__ bias, void* __restrict__ Cout, int M) {
  constexpr int K = HIDDEN;
  __shared__ __align__(16) unsigned short sW[NCOLS * K];

  // Cooperative stage of Wt into LDS (16B chunks).
  constexpr int CHUNKS = NCOLS * K * 2 / 16;
  {
    const uint4* g = (const uint4*)Wt;
    uint4* l = (uint4*)sW;
    #pragma unroll
    for (int c = 0; c < (CHUNKS + 255) / 256; ++c) {
      int idx = c * 256 + threadIdx.x;
      if (CHUNKS % 256 == 0 || idx < CHUNKS) l[idx] = g[idx];
    }
  }
  __syncthreads();

  const int lane = threadIdx.x & 31;
  const int wave = threadIdx.x >> 5;
  const int r0   = (blockIdx.x * 8 + wave) * 16;
  if (r0 >= M) return;                      // wave-uniform; after the barrier

  const int mA  = lane & 15;
  const int kh  = lane >> 4;                // K-half held by this lane
  const int row = r0 + mA;                  // always < M (M % 16 == 0)

  // A fragments: 2 x b128 global loads each (contiguous 8 bf16 per load).
  Frag a[4];
  const uint4* arow = (const uint4*)(A + (long)row * K);    // 16 uint4 per row
  #pragma unroll
  for (int kt = 0; kt < 4; ++kt) {
    a[kt].q[0] = arow[kt * 4 + kh];
    a[kt].q[1] = arow[kt * 4 + kh + 2];
  }

  const int nB = lane & 15;
  const int mhBase = kh * 8;                // C/D layout: lanes16-31 hold M+8
  #pragma unroll
  for (int ct = 0; ct < NCOLS / 16; ++ct) {
    const int cb = ct * 16;
    float cinit = HASBIAS ? bias[cb + nB] : 0.f;
    v8f c;
    #pragma unroll
    for (int i = 0; i < 8; ++i) c[i] = cinit;

    const uint4* brow = (const uint4*)(sW + (cb + nB) * K); // LDS, b128 reads
    #pragma unroll
    for (int kt = 0; kt < 4; ++kt) {
      Frag b;
      b.q[0] = brow[kt * 4 + kh];
      b.q[1] = brow[kt * 4 + kh + 2];
      c = __builtin_amdgcn_wmma_f32_16x16x32_bf16(
              false, a[kt].v, false, b.v, (short)0, c, false, false);
    }

    // Unconditional, wave-coalesced stores; rows NCOLS elements apart share
    // one address register via immediate offsets.
    if (BF16OUT) {
      __bf16* cp = (__bf16*)Cout + (long)(r0 + mhBase) * NCOLS + cb + nB;
      #pragma unroll
      for (int i = 0; i < 8; ++i) {
        float val = c[i];
        if (RELU) val = val > 0.f ? val : 0.f;
        cp[(long)i * NCOLS] = (__bf16)val;
      }
    } else {
      float* cp = (float*)Cout + (long)(r0 + mhBase) * NCOLS + cb + nB;
      #pragma unroll
      for (int i = 0; i < 8; ++i) {
        float val = c[i];
        if (RELU) val = val > 0.f ? val : 0.f;
        cp[(long)i * NCOLS] = val;
      }
    }
  }
}

// ---------------------------------------------------------------------------
// Sparse / elementwise kernels
// ---------------------------------------------------------------------------
__global__ void k_deg_init(float* deg) {
  int i = blockIdx.x * 256 + threadIdx.x;
  if (i < N_NODES) deg[i] = 1.0f;           // self-loop contribution
}

__global__ void k_deg_acc(const int* __restrict__ dst, float* __restrict__ deg) {
  int e = blockIdx.x * 256 + threadIdx.x;
  if (e < N_EDGES) atomicAdd(&deg[dst[e]], 1.0f);
}

__global__ void k_rsqrt(float* d) {
  int i = blockIdx.x * 256 + threadIdx.x;
  if (i < N_NODES) d[i] = rsqrtf(d[i]);
}

// Transpose + convert weights: Wt[n][k] = bf16(W[k][n]).
__global__ void k_cvt_w(const float* __restrict__ W, __bf16* __restrict__ Wt,
                        int Ncols) {
  int t = blockIdx.x * 256 + threadIdx.x;
  if (t >= HIDDEN * Ncols) return;
  int n = t / HIDDEN, k = t % HIDDEN;
  Wt[n * HIDDEN + k] = (__bf16)W[k * Ncols + n];
}

// Embedding gather, converted straight to bf16 (8 elems / thread).
__global__ void k_embed(const int* __restrict__ x, const float* __restrict__ emb,
                        __bf16* __restrict__ hbf) {
  int t = blockIdx.x * 256 + threadIdx.x;
  if (t >= N_NODES * 16) return;
  int node = t >> 4;
  int j = (t & 15) * 8;
  const float4* s = (const float4*)(emb + (long)x[node] * HIDDEN + j);
  float4 v0 = s[0], v1 = s[1];
  Pack8 p;
  p.b[0] = (__bf16)v0.x; p.b[1] = (__bf16)v0.y; p.b[2] = (__bf16)v0.z; p.b[3] = (__bf16)v0.w;
  p.b[4] = (__bf16)v1.x; p.b[5] = (__bf16)v1.y; p.b[6] = (__bf16)v1.z; p.b[7] = (__bf16)v1.w;
  *(uint4*)(hbf + (long)node * HIDDEN + j) = p.q;
}

// agg = bias + dinv[i]^2 * t[i]   (self-loop term, also initializes agg)
__global__ void k_agg_init(const float* __restrict__ t, const float* __restrict__ dinv,
                           const float* __restrict__ bias, float* __restrict__ agg) {
  int id = blockIdx.x * 256 + threadIdx.x;
  if (id >= N_NODES * 32) return;
  int i = id >> 5;
  int j = (id & 31) * 4;
  float c = dinv[i] * dinv[i];
  float4 v = *(const float4*)(t + (long)i * HIDDEN + j);
  float4 o;
  o.x = bias[j + 0] + c * v.x;
  o.y = bias[j + 1] + c * v.y;
  o.z = bias[j + 2] + c * v.z;
  o.w = bias[j + 3] + c * v.w;
  *(float4*)(agg + (long)i * HIDDEN + j) = o;
}

// One wave per edge; each lane scatters 4 floats with f32 atomics (L2-resident).
__global__ void k_agg_edges(const int* __restrict__ src, const int* __restrict__ dst,
                            const float* __restrict__ dinv, const float* __restrict__ t,
                            float* __restrict__ agg) {
  long tid = (long)blockIdx.x * 256 + threadIdx.x;
  int e = (int)(tid >> 5);
  if (e >= N_EDGES) return;
  int lane = (int)(tid & 31);
  int j = lane * 4;
  int s = src[e], d = dst[e];
  float c = dinv[s] * dinv[d];
  float4 v = *(const float4*)(t + (long)s * HIDDEN + j);
  float* outp = agg + (long)d * HIDDEN + j;
  atomicAdd(outp + 0, c * v.x);
  atomicAdd(outp + 1, c * v.y);
  atomicAdd(outp + 2, c * v.z);
  atomicAdd(outp + 3, c * v.w);
}

// ReLU + convert to bf16 staging for the next GEMM (8 elems / thread).
__global__ void k_relu_cvt(const float* __restrict__ agg, __bf16* __restrict__ hbf) {
  int t = blockIdx.x * 256 + threadIdx.x;
  if (t >= N_NODES * 16) return;
  int node = t >> 4;
  int j = (t & 15) * 8;
  const float4* s = (const float4*)(agg + (long)node * HIDDEN + j);
  float4 v0 = s[0], v1 = s[1];
  Pack8 p;
  p.b[0] = (__bf16)(v0.x > 0.f ? v0.x : 0.f);
  p.b[1] = (__bf16)(v0.y > 0.f ? v0.y : 0.f);
  p.b[2] = (__bf16)(v0.z > 0.f ? v0.z : 0.f);
  p.b[3] = (__bf16)(v0.w > 0.f ? v0.w : 0.f);
  p.b[4] = (__bf16)(v1.x > 0.f ? v1.x : 0.f);
  p.b[5] = (__bf16)(v1.y > 0.f ? v1.y : 0.f);
  p.b[6] = (__bf16)(v1.z > 0.f ? v1.z : 0.f);
  p.b[7] = (__bf16)(v1.w > 0.f ? v1.w : 0.f);
  *(uint4*)(hbf + (long)node * HIDDEN + j) = p.q;
}

// One block per subgraph, thread j handles feature j; masked mean over <=64 nodes.
__global__ __launch_bounds__(HIDDEN)
void k_pool(const int* __restrict__ subg, const __bf16* __restrict__ hbf,
            __bf16* __restrict__ pooled) {
  int s = blockIdx.x;
  int j = threadIdx.x;
  float sum = 0.f;
  int cnt = 0;
  #pragma unroll 4
  for (int m = 0; m < MAX_SUBG; ++m) {
    int idx = subg[s * MAX_SUBG + m];
    if (idx >= 0) { sum += (float)hbf[(long)idx * HIDDEN + j]; cnt++; }
  }
  pooled[s * HIDDEN + j] = (__bf16)(sum / (float)(cnt > 0 ? cnt : 1));
}

// ---------------------------------------------------------------------------
extern "C" void kernel_launch(void* const* d_in, const int* in_sizes, int n_in,
                              void* d_out, int out_size, void* d_ws, size_t ws_size,
                              hipStream_t stream) {
  const int*   x    = (const int*)d_in[0];
  const int*   eidx = (const int*)d_in[1];
  const int*   src  = eidx;
  const int*   dst  = eidx + N_EDGES;
  // d_in[2] = edge_attr (unused by reference)
  const int*   subg = (const int*)d_in[3];
  const float* emb  = (const float*)d_in[4];
  const float* W[3] = {(const float*)d_in[5], (const float*)d_in[7], (const float*)d_in[9]};
  const float* b[3] = {(const float*)d_in[6], (const float*)d_in[8], (const float*)d_in[10]};
  const float* mw1  = (const float*)d_in[11];
  const float* mb1  = (const float*)d_in[12];
  const float* mw2  = (const float*)d_in[13];
  const float* mb2  = (const float*)d_in[14];
  float* out = (float*)d_out;

  auto align256 = [](size_t v) { return (v + 255) & ~size_t(255); };
  char* ws = (char*)d_ws;
  size_t off = 0;
  float*  dinv   = (float*)(ws + off);  off += align256((size_t)N_NODES * 4);
  float*  bufA   = (float*)(ws + off);  off += align256((size_t)N_NODES * HIDDEN * 4);
  float*  bufB   = (float*)(ws + off);  off += align256((size_t)N_NODES * HIDDEN * 4);
  __bf16* hbf    = (__bf16*)(ws + off); off += align256((size_t)N_NODES * HIDDEN * 2);
  __bf16* wt[3];
  for (int l = 0; l < 3; ++l) { wt[l] = (__bf16*)(ws + off); off += align256((size_t)HIDDEN * HIDDEN * 2); }
  __bf16* wtm1   = (__bf16*)(ws + off); off += align256((size_t)HIDDEN * HIDDEN * 2);
  __bf16* wtm2   = (__bf16*)(ws + off); off += align256((size_t)HIDDEN * OUT_DIM * 2);
  __bf16* pooled = (__bf16*)(ws + off); off += align256((size_t)N_SUBG * HIDDEN * 2);
  __bf16* hid    = (__bf16*)(ws + off); off += align256((size_t)N_SUBG * HIDDEN * 2);

  const int TPB = 256;
  const int nodeVecBlocks  = (N_NODES * 32 + TPB - 1) / TPB;
  const int nodeVec8Blocks = (N_NODES * 16 + TPB - 1) / TPB;

  // 1) symmetric-norm degree factors
  k_deg_init<<<(N_NODES + TPB - 1) / TPB, TPB, 0, stream>>>(dinv);
  k_deg_acc <<<(N_EDGES + TPB - 1) / TPB, TPB, 0, stream>>>(dst, dinv);
  k_rsqrt   <<<(N_NODES + TPB - 1) / TPB, TPB, 0, stream>>>(dinv);

  // 2) weight transpose+convert (tiny)
  for (int l = 0; l < 3; ++l)
    k_cvt_w<<<(HIDDEN * HIDDEN + TPB - 1) / TPB, TPB, 0, stream>>>(W[l], wt[l], HIDDEN);
  k_cvt_w<<<(HIDDEN * HIDDEN + TPB - 1) / TPB, TPB, 0, stream>>>(mw1, wtm1, HIDDEN);
  k_cvt_w<<<(HIDDEN * OUT_DIM + TPB - 1) / TPB, TPB, 0, stream>>>(mw2, wtm2, OUT_DIM);

  // 3) embedding lookup -> hbf (bf16)
  k_embed<<<nodeVec8Blocks, TPB, 0, stream>>>(x, emb, hbf);

  // 4) three GCN layers: hbf -> bufB (t = h@W, f32) -> bufA (agg) -> hbf
  const int gemmBlocksN = ((N_NODES + 15) / 16 + 7) / 8;     // 782
  const int edgeBlocks  = (int)(((long)N_EDGES * 32 + TPB - 1) / TPB);
  for (int l = 0; l < 3; ++l) {
    gemm_wmma_bf16<HIDDEN, false, false, false>
        <<<gemmBlocksN, TPB, 0, stream>>>(hbf, wt[l], nullptr, bufB, N_NODES);
    k_agg_init <<<nodeVecBlocks, TPB, 0, stream>>>(bufB, dinv, b[l], bufA);
    k_agg_edges<<<edgeBlocks, TPB, 0, stream>>>(src, dst, dinv, bufB, bufA);
    k_relu_cvt <<<nodeVec8Blocks, TPB, 0, stream>>>(bufA, hbf);
  }

  // 5) masked mean pooling over subgraphs (bf16 out for the MLP head)
  k_pool<<<N_SUBG, HIDDEN, 0, stream>>>(subg, hbf, pooled);

  // 6) MLP head (WMMA, fused bias / ReLU; bf16 chaining)
  const int gemmBlocksS = ((N_SUBG + 15) / 16 + 7) / 8;      // 16
  gemm_wmma_bf16<HIDDEN, true, true, true>
      <<<gemmBlocksS, TPB, 0, stream>>>(pooled, wtm1, mb1, hid, N_SUBG);
  gemm_wmma_bf16<OUT_DIM, false, true, false>
      <<<gemmBlocksS, TPB, 0, stream>>>(hid, wtm2, mb2, out, N_SUBG);
}
